// HGTBackbone_11914239279717
// MI455X (gfx1250) — matrix operand order repaired and checked by reading, not compiled, not run on hardware
//
#include <hip/hip_runtime.h>
#include <cstring>

// ---------------------------------------------------------------------------
// HGT backbone for MI455X (gfx1250, wave32).
// Dense GEMMs (kqv, per-head relation transforms, output proj) run on the
// WMMA path: v_wmma_f32_16x16x32_bf16 (bf16 A/B, f32 accum), with the B panel
// staged once per block into LDS (bf16, transposed) and the K-loop fully
// unrolled. Irregular segment-softmax / scatter aggregation uses L2 atomics.
// ---------------------------------------------------------------------------

typedef __attribute__((ext_vector_type(16))) __bf16 v16bf;
typedef __attribute__((ext_vector_type(8)))  float  v8f;

static constexpr int NP = 40000;   // paper nodes
static constexpr int NA = 25000;   // author nodes
static constexpr int EW = 100000;  // writes: author -> paper
static constexpr int ER = 100000;  // rev:    paper  -> author
static constexpr int EC = 150000;  // cites:  paper  -> paper
static constexpr int CCH = 128;    // channels
static constexpr int HH  = 4;      // heads
static constexpr int DD  = 32;     // dim per head

// ---------------- order-preserving float <-> u32 encoding -----------------
__device__ __forceinline__ unsigned f2key(float f) {
    unsigned u = __float_as_uint(f);
    return (u & 0x80000000u) ? ~u : (u | 0x80000000u);
}
__device__ __forceinline__ float key2f(unsigned k) {
    return (k & 0x80000000u) ? __uint_as_float(k ^ 0x80000000u)
                             : __uint_as_float(~k);
}
static unsigned f2key_host(float f) {
    unsigned u; memcpy(&u, &f, 4);
    return (u & 0x80000000u) ? ~u : (u | 0x80000000u);
}

// ---------------------- WMMA bf16 GEMM: C = A*B (+bias) --------------------
// A: M x KT (row stride lda, fp32), B: KT x N (row stride ldb, fp32),
// C: M x N (row stride ldc, fp32). KT in {32,128}; N % 32 == 0; M ragged.
// Block = 4 waves; each wave owns a 16x32 output tile (2 WMMA tiles, shared
// A fragment). B panel (KT x 32) staged once per block into LDS as bf16,
// stored transposed so each lane's fragment is two runs of 8 contiguous bf16 (b128 ds each).
template<int KT>
__global__ __launch_bounds__(128)
void wmma_gemm_bf16(const float* __restrict__ A, int lda,
                    const float* __restrict__ B, int ldb,
                    float* __restrict__ C, int ldc,
                    const float* __restrict__ bias,
                    int M, int N)
{
    __shared__ __bf16 Bs[32][KT + 8];           // [n][k], padded rows

    const int lane    = threadIdx.x;            // 0..31
    const int wv      = threadIdx.y;            // 0..3
    const int tid     = wv * 32 + lane;
    const int colbase = blockIdx.x * 32;
    const int row0    = (blockIdx.y * 4 + wv) * 16;

    // ---- cooperative stage of B panel (coalesced: 32 consecutive n per k) ----
    #pragma unroll 4
    for (int e = tid; e < KT * 32; e += 128) {
        int k = e >> 5, n = e & 31;
        Bs[n][k] = (__bf16)B[(long)k * ldb + colbase + n];
    }
    __syncthreads();

    if (row0 < M) {                             // wave-uniform guard
        const int half = (lane >> 4) & 1;
        const int lm   = lane & 15;
        int arow = row0 + lm;
        if (arow >= M) arow = M - 1;            // clamp: garbage rows discarded at store
        const float* __restrict__ Arow = A + (long)arow * lda;

        union { v16bf v; __bf16 e[16]; } af, bf0, bf1;
        union { v8f   v; float  e[8];  } c0, c1;
        #pragma unroll
        for (int i = 0; i < 8; ++i) { c0.e[i] = 0.0f; c1.e[i] = 0.0f; }

        #pragma unroll
        for (int k0 = 0; k0 < KT; k0 += 32) {
            // A fragment (16x32 bf16): elem j -> K = k0 + half*8 + (j&7) + (j>>3)*16
            const int kb = k0 + half * 8;
            float4 a0 = *(const float4*)(Arow + kb);
            float4 a1 = *(const float4*)(Arow + kb + 4);
            float4 a2 = *(const float4*)(Arow + kb + 16);
            float4 a3 = *(const float4*)(Arow + kb + 20);
            af.e[0]  = (__bf16)a0.x; af.e[1]  = (__bf16)a0.y;
            af.e[2]  = (__bf16)a0.z; af.e[3]  = (__bf16)a0.w;
            af.e[4]  = (__bf16)a1.x; af.e[5]  = (__bf16)a1.y;
            af.e[6]  = (__bf16)a1.z; af.e[7]  = (__bf16)a1.w;
            af.e[8]  = (__bf16)a2.x; af.e[9]  = (__bf16)a2.y;
            af.e[10] = (__bf16)a2.z; af.e[11] = (__bf16)a2.w;
            af.e[12] = (__bf16)a3.x; af.e[13] = (__bf16)a3.y;
            af.e[14] = (__bf16)a3.z; af.e[15] = (__bf16)a3.w;

            // B fragments (32x16 bf16): elem j -> K = k0 + half*8 + (j&7) + (j>>3)*16
            const int kc = k0 + half * 8;
            #pragma unroll
            for (int j = 0; j < 16; ++j) {
                bf0.e[j] = Bs[lm][kc + (j & 7) + (j >> 3) * 16];
                bf1.e[j] = Bs[16 + lm][kc + (j & 7) + (j >> 3) * 16];
            }
            c0.v = __builtin_amdgcn_wmma_f32_16x16x32_bf16(
                false, af.v, false, bf0.v, (short)0, c0.v, false, false);
            c1.v = __builtin_amdgcn_wmma_f32_16x16x32_bf16(
                false, af.v, false, bf1.v, (short)0, c1.v, false, false);
        }

        #pragma unroll
        for (int i = 0; i < 8; ++i) {
            int m = row0 + i + half * 8;
            if (m < M) {
                int n0 = colbase + lm, n1 = n0 + 16;
                float r0 = c0.e[i], r1 = c1.e[i];
                if (bias) { r0 += bias[n0]; r1 += bias[n1]; }
                C[(long)m * ldc + n0] = r0;
                C[(long)m * ldc + n1] = r1;
            }
        }
    }
}

// ------------------------------ fills --------------------------------------
__global__ void fill_u32(unsigned* p, unsigned v, int n) {
    int i = blockIdx.x * blockDim.x + threadIdx.x;
    if (i < n) p[i] = v;
}
__global__ void fill_f32(float* p, float v, int n) {
    int i = blockIdx.x * blockDim.x + threadIdx.x;
    if (i < n) p[i] = v;
}

// --------------------- edge attention: score + seg-max ---------------------
// thread per (edge, head); float4-vectorized 32-dot
__global__ void edge_score(const int* __restrict__ ei, int E,
                           const float* __restrict__ q, int ldq,
                           const float* __restrict__ kt,
                           const float* __restrict__ prel,
                           float* __restrict__ sc,
                           unsigned* __restrict__ mkey)
{
    int idx = blockIdx.x * blockDim.x + threadIdx.x;
    if (idx >= E * HH) return;
    int e = idx >> 2, h = idx & 3;
    int src = ei[e], dst = ei[E + e];
    const float4* q4 = (const float4*)(q  + (long)dst * ldq + h * DD);
    const float4* k4 = (const float4*)(kt + (long)src * CCH + h * DD);
    float dot = 0.0f;
    #pragma unroll
    for (int d = 0; d < 8; ++d) {
        float4 a = q4[d], b = k4[d];
        dot += a.x * b.x + a.y * b.y + a.z * b.z + a.w * b.w;
    }
    float s = dot * prel[h] * 0.17677669529663687f;  // 1/sqrt(32)
    sc[idx] = s;
    atomicMax(&mkey[dst * HH + h], f2key(s));
}

// --------------------- edge attention: exp + seg-sum -----------------------
__global__ void edge_exp(const int* __restrict__ ei, int E,
                         float* __restrict__ sc,
                         const unsigned* __restrict__ mkey,
                         float* __restrict__ ssum)
{
    int idx = blockIdx.x * blockDim.x + threadIdx.x;
    if (idx >= E * HH) return;
    int e = idx >> 2, h = idx & 3;
    int dst = ei[E + e];
    float m  = key2f(mkey[dst * HH + h]);
    float ev = expf(sc[idx] - m);
    sc[idx]  = ev;
    atomicAdd(&ssum[dst * HH + h], ev);
}

// ------------------- weighted scatter-add of values ------------------------
// thread per (edge, channel)
__global__ void edge_agg(const int* __restrict__ ei, int E,
                         const float* __restrict__ sc,
                         const float* __restrict__ ssum,
                         const float* __restrict__ vt,
                         float* __restrict__ agg)
{
    long idx = (long)blockIdx.x * blockDim.x + threadIdx.x;
    if (idx >= (long)E * CCH) return;
    int e = (int)(idx >> 7);
    int c = (int)(idx & 127);
    int h = c >> 5;
    int src = ei[e], dst = ei[E + e];
    float a = sc[(e << 2) + h] / (ssum[dst * HH + h] + 1e-16f);
    atomicAdd(&agg[(long)dst * CCH + c], vt[(long)src * CCH + c] * a);
}

// ------------------------------ gelu (exact) -------------------------------
__global__ void gelu_inplace(float* __restrict__ p, long n) {
    long i = (long)blockIdx.x * blockDim.x + threadIdx.x;
    if (i < n) {
        float x = p[i];
        p[i] = 0.5f * x * (1.0f + erff(x * 0.70710678118654752f));
    }
}

// ------------------- gated residual + relu (in place on x) -----------------
__global__ void residual_relu(const float* __restrict__ o,
                              float* __restrict__ x,
                              const float* __restrict__ skip, long n)
{
    long i = (long)blockIdx.x * blockDim.x + threadIdx.x;
    if (i < n) {
        float al = 1.0f / (1.0f + expf(-skip[0]));
        float v  = al * o[i] + (1.0f - al) * x[i];
        x[i] = v > 0.0f ? v : 0.0f;
    }
}

// ===========================================================================
extern "C" void kernel_launch(void* const* d_in, const int* in_sizes, int n_in,
                              void* d_out, int out_size, void* d_ws, size_t ws_size,
                              hipStream_t stream)
{
    (void)in_sizes; (void)n_in; (void)out_size; (void)ws_size;

    const int* ei_w = (const int*)d_in[2];
    const int* ei_r = (const int*)d_in[3];
    const int* ei_c = (const int*)d_in[4];

    // -------- workspace carve-out (floats) --------
    float* w = (float*)d_ws;
    size_t off = 0;
    auto alloc = [&](size_t n) { float* p = w + off; off += n; return p; };
    float* xp     = alloc((size_t)NP * CCH);
    float* xa     = alloc((size_t)NA * CCH);
    float* kqv_p  = alloc((size_t)NP * 3 * CCH);
    float* kqv_a  = alloc((size_t)NA * 3 * CCH);
    float* kt_w   = alloc((size_t)NA * CCH);
    float* vt_w   = alloc((size_t)NA * CCH);
    float* kt_r   = alloc((size_t)NP * CCH);
    float* vt_r   = alloc((size_t)NP * CCH);
    float* kt_c   = alloc((size_t)NP * CCH);
    float* vt_c   = alloc((size_t)NP * CCH);
    unsigned* mkey_p = (unsigned*)alloc((size_t)NP * HH);
    float* ssum_p = alloc((size_t)NP * HH);
    unsigned* mkey_a = (unsigned*)alloc((size_t)NA * HH);
    float* ssum_a = alloc((size_t)NA * HH);
    float* agg_p  = alloc((size_t)NP * CCH);
    float* agg_a  = alloc((size_t)NA * CCH);
    float* sc_w   = alloc((size_t)EW * HH);
    float* sc_r   = alloc((size_t)ER * HH);
    float* sc_c   = alloc((size_t)EC * HH);
    float* ob_p   = kqv_p;   // kqv dead after edge passes -> reuse for out-proj
    float* ob_a   = kqv_a;

    hipMemcpyAsync(xp, d_in[0], (size_t)NP * CCH * 4, hipMemcpyDeviceToDevice, stream);
    hipMemcpyAsync(xa, d_in[1], (size_t)NA * CCH * 4, hipMemcpyDeviceToDevice, stream);

    const unsigned NEG_INF_KEY = f2key_host(-3.0e38f);

    auto gemm = [&](const float* A, int lda, const float* B, int ldb,
                    float* C, int ldc, const float* bias, int M, int N, int K) {
        dim3 grid(N / 32, (M + 63) / 64);
        if (K == 128)
            wmma_gemm_bf16<128><<<grid, dim3(32, 4), 0, stream>>>(A, lda, B, ldb, C, ldc, bias, M, N);
        else
            wmma_gemm_bf16<32><<<grid, dim3(32, 4), 0, stream>>>(A, lda, B, ldb, C, ldc, bias, M, N);
    };

    for (int l = 0; l < 2; ++l) {
        // params: JAX pytree order (sorted dict keys)
        const int b = 5 + 19 * l;
        const float* kqvW_a = (const float*)d_in[b + 0];
        const float* kqvB_a = (const float*)d_in[b + 1];
        const float* kqvW_p = (const float*)d_in[b + 2];
        const float* kqvB_p = (const float*)d_in[b + 3];
        const float* outW_a = (const float*)d_in[b + 4];
        const float* outB_a = (const float*)d_in[b + 5];
        const float* outW_p = (const float*)d_in[b + 6];
        const float* outB_p = (const float*)d_in[b + 7];
        const float* Ak_c   = (const float*)d_in[b + 8];
        const float* Av_c   = (const float*)d_in[b + 9];
        const float* pr_c   = (const float*)d_in[b + 10];
        const float* Ak_r   = (const float*)d_in[b + 11];
        const float* Av_r   = (const float*)d_in[b + 12];
        const float* pr_r   = (const float*)d_in[b + 13];
        const float* Ak_w   = (const float*)d_in[b + 14];
        const float* Av_w   = (const float*)d_in[b + 15];
        const float* pr_w   = (const float*)d_in[b + 16];
        const float* skip_a = (const float*)d_in[b + 17];
        const float* skip_p = (const float*)d_in[b + 18];

        // ---- kqv = x @ W + b ; row layout [k(128) | q(128) | v(128)] ----
        gemm(xp, CCH, kqvW_p, 3 * CCH, kqv_p, 3 * CCH, kqvB_p, NP, 3 * CCH, CCH);
        gemm(xa, CCH, kqvW_a, 3 * CCH, kqv_a, 3 * CCH, kqvB_a, NA, 3 * CCH, CCH);

        // ---- per-head relation transforms: kt/vt = k/v @ A[h] (32x32) ----
        for (int h = 0; h < HH; ++h) {
            // writes (source = author)
            gemm(kqv_a + 0       + h * DD, 3 * CCH, Ak_w + h * DD * DD, DD,
                 kt_w + h * DD, CCH, nullptr, NA, DD, DD);
            gemm(kqv_a + 2 * CCH + h * DD, 3 * CCH, Av_w + h * DD * DD, DD,
                 vt_w + h * DD, CCH, nullptr, NA, DD, DD);
            // rev (source = paper)
            gemm(kqv_p + 0       + h * DD, 3 * CCH, Ak_r + h * DD * DD, DD,
                 kt_r + h * DD, CCH, nullptr, NP, DD, DD);
            gemm(kqv_p + 2 * CCH + h * DD, 3 * CCH, Av_r + h * DD * DD, DD,
                 vt_r + h * DD, CCH, nullptr, NP, DD, DD);
            // cites (source = paper)
            gemm(kqv_p + 0       + h * DD, 3 * CCH, Ak_c + h * DD * DD, DD,
                 kt_c + h * DD, CCH, nullptr, NP, DD, DD);
            gemm(kqv_p + 2 * CCH + h * DD, 3 * CCH, Av_c + h * DD * DD, DD,
                 vt_c + h * DD, CCH, nullptr, NP, DD, DD);
        }

        // ---- init per-dest softmax state ----
        fill_u32<<<(NP * HH + 255) / 256, 256, 0, stream>>>(mkey_p, NEG_INF_KEY, NP * HH);
        fill_u32<<<(NA * HH + 255) / 256, 256, 0, stream>>>(mkey_a, NEG_INF_KEY, NA * HH);
        fill_f32<<<(NP * HH + 255) / 256, 256, 0, stream>>>(ssum_p, 0.0f, NP * HH);
        fill_f32<<<(NA * HH + 255) / 256, 256, 0, stream>>>(ssum_a, 0.0f, NA * HH);
        fill_f32<<<(NP * CCH + 255) / 256, 256, 0, stream>>>(agg_p, 0.0f, NP * CCH);
        fill_f32<<<(NA * CCH + 255) / 256, 256, 0, stream>>>(agg_a, 0.0f, NA * CCH);

        const float* q_p = kqv_p + CCH;   // q block, row stride 384
        const float* q_a = kqv_a + CCH;

        // ---- scores + segment max ----
        edge_score<<<(EW * HH + 255) / 256, 256, 0, stream>>>(ei_w, EW, q_p, 3 * CCH, kt_w, pr_w, sc_w, mkey_p);
        edge_score<<<(EC * HH + 255) / 256, 256, 0, stream>>>(ei_c, EC, q_p, 3 * CCH, kt_c, pr_c, sc_c, mkey_p);
        edge_score<<<(ER * HH + 255) / 256, 256, 0, stream>>>(ei_r, ER, q_a, 3 * CCH, kt_r, pr_r, sc_r, mkey_a);

        // ---- exp + segment sum ----
        edge_exp<<<(EW * HH + 255) / 256, 256, 0, stream>>>(ei_w, EW, sc_w, mkey_p, ssum_p);
        edge_exp<<<(EC * HH + 255) / 256, 256, 0, stream>>>(ei_c, EC, sc_c, mkey_p, ssum_p);
        edge_exp<<<(ER * HH + 255) / 256, 256, 0, stream>>>(ei_r, ER, sc_r, mkey_a, ssum_a);

        // ---- weighted scatter aggregation ----
        edge_agg<<<(int)(((long)EW * CCH + 255) / 256), 256, 0, stream>>>(ei_w, EW, sc_w, ssum_p, vt_w, agg_p);
        edge_agg<<<(int)(((long)EC * CCH + 255) / 256), 256, 0, stream>>>(ei_c, EC, sc_c, ssum_p, vt_c, agg_p);
        edge_agg<<<(int)(((long)ER * CCH + 255) / 256), 256, 0, stream>>>(ei_r, ER, sc_r, ssum_a, vt_r, agg_a);

        // ---- gelu -> output projection -> gated residual + relu ----
        gelu_inplace<<<(int)(((long)NP * CCH + 255) / 256), 256, 0, stream>>>(agg_p, (long)NP * CCH);
        gelu_inplace<<<(int)(((long)NA * CCH + 255) / 256), 256, 0, stream>>>(agg_a, (long)NA * CCH);

        gemm(agg_p, CCH, outW_p, CCH, ob_p, CCH, outB_p, NP, CCH, CCH);
        gemm(agg_a, CCH, outW_a, CCH, ob_a, CCH, outB_a, NA, CCH, CCH);

        residual_relu<<<(int)(((long)NP * CCH + 255) / 256), 256, 0, stream>>>(ob_p, xp, skip_p, (long)NP * CCH);
        residual_relu<<<(int)(((long)NA * CCH + 255) / 256), 256, 0, stream>>>(ob_a, xa, skip_a, (long)NA * CCH);
    }

    // outputs: (paper, author) concatenated
    float* outp = (float*)d_out;
    hipMemcpyAsync(outp,                    xp, (size_t)NP * CCH * 4, hipMemcpyDeviceToDevice, stream);
    hipMemcpyAsync(outp + (size_t)NP * CCH, xa, (size_t)NA * CCH * 4, hipMemcpyDeviceToDevice, stream);
}
